// PairEGNN_33981781246506
// MI455X (gfx1250) — matrix operand order repaired and checked by reading, not compiled, not run on hardware
//
#include <hip/hip_runtime.h>
#include <hip/hip_bf16.h>

// ---------------- problem constants ----------------
#define NL     4                 // layers
#define H      256               // hidden
#define D_IN   16
#define D_EDGE 4
#define NB     64                // graphs
#define NN     48                // nodes per graph
#define NNODE  (NB * NN)         // 3072
#define NE     (NB * NN * (NN-1))// 144384
#define D_E1   517               // 2H + 1 + D_EDGE
#define D_N1   528               // 2H + D_IN
#define KP     544               // padded K (multiple of 32) for first MLP layers

#define EM     64                // edges per block (4 M-subtiles of 16)
// dynamic LDS layout for the edge kernel
#define SM_EIN   (EM * KP * 2)           // 69632 B : e_in bf16 tile (later reused as f32 redbuf)
#define SM_MID   (EM * H * 2)            // 32768 B : layer-1 output bf16 tile
#define SM_TAIL  (EM * 16)               // 1024 B  : row/col/radial/mask
#define SMEM_EDGE (SM_EIN + SM_MID + SM_TAIL)

// ---------------- WMMA types ----------------
typedef __attribute__((ext_vector_type(16))) __bf16       v16bf;
typedef __attribute__((ext_vector_type(8)))  float        v8f;
typedef __attribute__((ext_vector_type(4)))  unsigned int v4u;

union ABFrag { v16bf v; v4u q[2]; };

__device__ __forceinline__ unsigned short f2bf(float f) {
  union { float f; unsigned u; } c; c.f = f;
  unsigned r = c.u + 0x7FFFu + ((c.u >> 16) & 1u);   // round-to-nearest-even
  return (unsigned short)(r >> 16);
}

__device__ __forceinline__ float silu(float x) {
  return x / (1.0f + __expf(-x));
}

// A-matrix fragment (16xK tile, row-major bf16 in LDS, stride strideK halfwords).
// ISA 16-bit A layout: lane m=l&15, half=l>>4; u[0..7]=K half*8..+7, u[8..15]=K 16+half*8..+7.
__device__ __forceinline__ v16bf load_a_lds(const unsigned short* sh, int strideK,
                                            int kstep, int lane) {
  int m = lane & 15, half = lane >> 4;
  const unsigned short* p = sh + m * strideK + kstep * 32 + half * 8;
  ABFrag f;
  f.q[0] = *(const v4u*)(p);
  f.q[1] = *(const v4u*)(p + 16);
  return f.v;
}

// B-matrix fragment from global weights stored [Ncols][K] bf16 (K contiguous).
// ISA 16-bit B layout: lane n=l&15 (column), half=l>>4; u[0..7]=K half*8..+7, u[8..15]=K 16+half*8..+7.
__device__ __forceinline__ v16bf load_b_glb(const unsigned short* __restrict__ w, int K,
                                            int nt, int kstep, int lane) {
  int n  = nt * 16 + (lane & 15);
  int kk = kstep * 32 + (lane >> 4) * 8;
  const v4u* p = (const v4u*)(w + n * K + kk);
  ABFrag f; f.q[0] = p[0]; f.q[1] = p[2];
  return f.v;
}

__device__ __forceinline__ v8f wmma_bf16(v16bf a, v16bf b, v8f c) {
  return __builtin_amdgcn_wmma_f32_16x16x32_bf16(false, a, false, b, (short)0, c,
                                                 false, false);
}

// ---------------- weight prep: f32 [L][Kin][H] -> bf16 [L][H][Kpad] (transposed+padded) ----
__global__ __launch_bounds__(256)
void wprep_kernel(const float* __restrict__ W, unsigned short* __restrict__ out,
                  int Kin, int Kpad) {
  int i = blockIdx.x * 256 + threadIdx.x;
  int total = NL * H * Kpad;
  if (i >= total) return;
  int l   = i / (H * Kpad);
  int rem = i - l * (H * Kpad);
  int n   = rem / Kpad;
  int k   = rem - n * Kpad;
  out[i] = (k < Kin) ? f2bf(W[((size_t)l * Kin + k) * H + n]) : (unsigned short)0;
}

// ---------------- embedding: h = h0 @ embW + b ----------------
__global__ __launch_bounds__(256)
void emb_kernel(const float* __restrict__ h0, const float* __restrict__ W,
                const float* __restrict__ b, float* __restrict__ h) {
  int i = blockIdx.x * 256 + threadIdx.x;
  if (i >= NNODE * H) return;
  int n = i >> 8, c = i & (H - 1);
  float acc = b[c];
  #pragma unroll
  for (int k = 0; k < D_IN; ++k) acc += h0[n * D_IN + k] * W[k * H + c];
  h[i] = acc;
}

// ---------------- per-layer prologue: hbf = bf16(h); agg = 0 ----------------
__global__ __launch_bounds__(256)
void cvt_zero_kernel(const float* __restrict__ h, unsigned short* __restrict__ hbf,
                     float* __restrict__ agg) {
  int i = blockIdx.x * 256 + threadIdx.x;
  if (i >= NNODE * H) return;
  hbf[i] = f2bf(h[i]);
  agg[i] = 0.0f;
}

__global__ __launch_bounds__(256)
void zero_kernel(float* __restrict__ p, int n) {
  int i = blockIdx.x * 256 + threadIdx.x;
  if (i < n) p[i] = 0.0f;
}

// ---------------- fused edge MLP + segment-sum ----------------
// One block = 64 edges (4 M-subtiles); 8 waves, each wave owns 2 of the 16 column tiles
// and all 4 M-subtiles, so every B fragment is reused 4x. Epilogue does an LDS
// segmented row-reduction before hitting global atomics (~20x fewer atomics).
__global__ __launch_bounds__(256, 2)
void edge_mlp_kernel(const unsigned short* __restrict__ hbf,
                     const float* __restrict__ x,
                     const int* __restrict__ edges,        // [2][NE]
                     const float* __restrict__ edge_attr,  // [NE][4]
                     const float* __restrict__ edge_mask,  // [NE]
                     const unsigned short* __restrict__ We1, const float* __restrict__ be1,
                     const unsigned short* __restrict__ We2, const float* __restrict__ be2,
                     float* __restrict__ agg) {
  extern __shared__ char smem[];
  unsigned short* sh_ein = (unsigned short*)smem;              // EM x KP bf16
  float*          redbuf = (float*)smem;                       // EM x H f32 (aliases sh_ein)
  unsigned short* sh_mid = (unsigned short*)(smem + SM_EIN);   // EM x H bf16
  char*           tail   = smem + SM_EIN + SM_MID;
  int*   sh_row = (int*)  (tail);
  int*   sh_col = (int*)  (tail + EM * 4);
  float* sh_rad = (float*)(tail + EM * 8);
  float* sh_msk = (float*)(tail + EM * 12);

  const int tid = threadIdx.x;
  const int e0  = blockIdx.x * EM;

  if (tid < EM) {
    int e = e0 + tid;
    int r = edges[e];
    int c = edges[NE + e];
    sh_row[tid] = r;
    sh_col[tid] = c;
    float dx = x[r * 3 + 0] - x[c * 3 + 0];
    float dy = x[r * 3 + 1] - x[c * 3 + 1];
    float dz = x[r * 3 + 2] - x[c * 3 + 2];
    sh_rad[tid] = dx * dx + dy * dy + dz * dz;
    sh_msk[tid] = edge_mask[e];
  }
  __syncthreads();

  // stage e_in = [h[row] | h[col] | radial | edge_attr | 0-pad] as bf16
  // vectorized 16B copies for the two h gathers (64 chunks of 8 halfwords per row)
  for (int j = tid; j < EM * 64; j += 256) {
    int r = j >> 6;
    int c = j & 63;
    const v4u* src = (c < 32) ? ((const v4u*)(hbf + sh_row[r] * H) + c)
                              : ((const v4u*)(hbf + sh_col[r] * H) + (c - 32));
    *((v4u*)(sh_ein + r * KP) + c) = *src;
  }
  // tail features (k = 512..543): radial, edge_attr, zero pad
  for (int j = tid; j < EM * 32; j += 256) {
    int r = j >> 5;
    int k = 2 * H + (j & 31);
    unsigned short v;
    if      (k == 2 * H) v = f2bf(sh_rad[r]);
    else if (k < D_E1)   v = f2bf(edge_attr[(e0 + r) * D_EDGE + (k - 2 * H - 1)]);
    else                 v = 0;
    sh_ein[r * KP + k] = v;
  }
  __syncthreads();

  const int lane = tid & 31;
  const int wave = tid >> 5;
  const int half = lane >> 4;
  const int nt0  = wave * 2;
  const int nt1  = wave * 2 + 1;
  const int nc0  = nt0 * 16 + (lane & 15);
  const int nc1  = nc0 + 16;

  // ---- layer 1: [64 x 544] @ [544 x 256] -> SiLU -> bf16 LDS tile ----
  {
    v8f acc[2][4];
    float b0 = be1[nc0], b1 = be1[nc1];
    #pragma unroll
    for (int mt = 0; mt < 4; ++mt)
      #pragma unroll
      for (int v = 0; v < 8; ++v) { acc[0][mt][v] = b0; acc[1][mt][v] = b1; }

    #pragma unroll
    for (int ks = 0; ks < KP / 32; ++ks) {
      v16bf bf0 = load_b_glb(We1, KP, nt0, ks, lane);
      v16bf bf1 = load_b_glb(We1, KP, nt1, ks, lane);
      #pragma unroll
      for (int mt = 0; mt < 4; ++mt) {
        v16bf a = load_a_lds(sh_ein + mt * 16 * KP, KP, ks, lane);
        acc[0][mt] = wmma_bf16(a, bf0, acc[0][mt]);
        acc[1][mt] = wmma_bf16(a, bf1, acc[1][mt]);
      }
    }
    #pragma unroll
    for (int mt = 0; mt < 4; ++mt)
      #pragma unroll
      for (int v = 0; v < 8; ++v) {
        int m = mt * 16 + v + 8 * half;            // C/D layout row
        sh_mid[m * H + nc0] = f2bf(silu(acc[0][mt][v]));
        sh_mid[m * H + nc1] = f2bf(silu(acc[1][mt][v]));
      }
  }
  __syncthreads();   // sh_mid ready; sh_ein region now free -> becomes redbuf

  // ---- layer 2: [64 x 256] @ [256 x 256] -> SiLU * mask -> redbuf ----
  {
    v8f acc[2][4];
    float b0 = be2[nc0], b1 = be2[nc1];
    #pragma unroll
    for (int mt = 0; mt < 4; ++mt)
      #pragma unroll
      for (int v = 0; v < 8; ++v) { acc[0][mt][v] = b0; acc[1][mt][v] = b1; }

    #pragma unroll
    for (int ks = 0; ks < H / 32; ++ks) {
      v16bf bf0 = load_b_glb(We2, H, nt0, ks, lane);
      v16bf bf1 = load_b_glb(We2, H, nt1, ks, lane);
      #pragma unroll
      for (int mt = 0; mt < 4; ++mt) {
        v16bf a = load_a_lds(sh_mid + mt * 16 * H, H, ks, lane);
        acc[0][mt] = wmma_bf16(a, bf0, acc[0][mt]);
        acc[1][mt] = wmma_bf16(a, bf1, acc[1][mt]);
      }
    }
    #pragma unroll
    for (int mt = 0; mt < 4; ++mt)
      #pragma unroll
      for (int v = 0; v < 8; ++v) {
        int m = mt * 16 + v + 8 * half;
        redbuf[m * H + nc0] = silu(acc[0][mt][v]) * sh_msk[m];
        redbuf[m * H + nc1] = silu(acc[1][mt][v]) * sh_msk[m];
      }
  }
  __syncthreads();

  // ---- segmented row reduction: one column per thread, one atomic per row-run ----
  {
    int c = tid;                                   // 256 threads == H columns
    float s = 0.0f;
    int prev = sh_row[0];
    #pragma unroll 8
    for (int m = 0; m < EM; ++m) {
      int r = sh_row[m];
      if (r != prev) {
        atomicAdd(&agg[prev * H + c], s);
        s = 0.0f;
        prev = r;
      }
      s += redbuf[m * H + c];
    }
    atomicAdd(&agg[prev * H + c], s);
  }
}

// ---------------- fused node MLP (+ residual) ----------------
__global__ __launch_bounds__(256, 2)
void node_mlp_kernel(const unsigned short* __restrict__ hbf,
                     const float* __restrict__ agg,
                     const float* __restrict__ h0,
                     const unsigned short* __restrict__ Wn1, const float* __restrict__ bn1,
                     const unsigned short* __restrict__ Wn2, const float* __restrict__ bn2,
                     float* __restrict__ h) {
  __shared__ unsigned short sh_nin[16 * KP];
  __shared__ unsigned short sh_mid[16 * H];

  const int tid = threadIdx.x;
  const int n0  = blockIdx.x * 16;

  // stage n_in = [h | agg | h0 | 0-pad] as bf16
  for (int j = tid; j < 16 * KP; j += 256) {
    int r = j / KP;
    int k = j - r * KP;
    int n = n0 + r;
    unsigned short v;
    if      (k < H)     v = hbf[n * H + k];
    else if (k < 2 * H) v = f2bf(agg[n * H + (k - H)]);
    else if (k < D_N1)  v = f2bf(h0[n * D_IN + (k - 2 * H)]);
    else                v = 0;
    sh_nin[j] = v;
  }
  __syncthreads();

  const int lane = tid & 31;
  const int wave = tid >> 5;
  const int half = lane >> 4;

  #pragma unroll
  for (int t = 0; t < 2; ++t) {
    int nt = wave * 2 + t;
    int nc = nt * 16 + (lane & 15);
    float bb = bn1[nc];
    v8f acc;
    #pragma unroll
    for (int v = 0; v < 8; ++v) acc[v] = bb;
    #pragma unroll
    for (int ks = 0; ks < KP / 32; ++ks) {
      v16bf a = load_a_lds(sh_nin, KP, ks, lane);
      v16bf b = load_b_glb(Wn1, KP, nt, ks, lane);
      acc = wmma_bf16(a, b, acc);
    }
    #pragma unroll
    for (int v = 0; v < 8; ++v) {
      int m = v + 8 * half;
      sh_mid[m * H + nc] = f2bf(silu(acc[v]));
    }
  }
  __syncthreads();

  #pragma unroll
  for (int t = 0; t < 2; ++t) {
    int nt = wave * 2 + t;
    int nc = nt * 16 + (lane & 15);
    float bb = bn2[nc];
    v8f acc;
    #pragma unroll
    for (int v = 0; v < 8; ++v) acc[v] = bb;
    #pragma unroll
    for (int ks = 0; ks < H / 32; ++ks) {
      v16bf a = load_a_lds(sh_mid, H, ks, lane);
      v16bf b = load_b_glb(Wn2, H, nt, ks, lane);
      acc = wmma_bf16(a, b, acc);
    }
    #pragma unroll
    for (int v = 0; v < 8; ++v) {          // second node layer is linear; residual h += out
      int m = v + 8 * half;
      int n = n0 + m;
      h[n * H + nc] += acc[v];
    }
  }
}

// ---------------- node decoder stage 1: hd = silu(h @ W1 + b1) ----------------
__global__ __launch_bounds__(256)
void dec1_kernel(const float* __restrict__ h, const float* __restrict__ W1,
                 const float* __restrict__ b1, float* __restrict__ hd) {
  int i = blockIdx.x * 256 + threadIdx.x;
  if (i >= NNODE * H) return;
  int n = i >> 8, c = i & (H - 1);
  float acc = b1[c];
  for (int k = 0; k < H; ++k) acc += h[n * H + k] * W1[k * H + c];
  hd[i] = silu(acc);
}

// ---------------- node decoder stage 2 + masked per-graph sum ----------------
__global__ __launch_bounds__(256)
void dec2_kernel(const float* __restrict__ hd, const float* __restrict__ W2,
                 const float* __restrict__ b2, const float* __restrict__ mask,
                 float* __restrict__ gsum) {
  int i = blockIdx.x * 256 + threadIdx.x;
  if (i >= NNODE * H) return;
  int n = i >> 8, c = i & (H - 1);
  float acc = b2[c];
  for (int k = 0; k < H; ++k) acc += hd[n * H + k] * W2[k * H + c];
  acc *= mask[n];
  atomicAdd(&gsum[(n / NN) * H + c], acc);
}

// ---------------- graph decoder: pred = silu([gs|gt] @ W1 + b1) @ W2 + b2 ----------------
__global__ __launch_bounds__(256)
void graph_dec_kernel(const float* __restrict__ gs, const float* __restrict__ gt,
                      const float* __restrict__ W1, const float* __restrict__ b1,
                      const float* __restrict__ W2, const float* __restrict__ b2,
                      float* __restrict__ out) {
  __shared__ float u[2 * H];
  __shared__ float red[256];
  int b = blockIdx.x, tid = threadIdx.x;
  for (int c = tid; c < 2 * H; c += 256) {
    float acc = b1[c];
    for (int j = 0; j < H; ++j) acc += gs[b * H + j] * W1[j * (2 * H) + c];
    for (int j = 0; j < H; ++j) acc += gt[b * H + j] * W1[(H + j) * (2 * H) + c];
    u[c] = silu(acc);
  }
  __syncthreads();
  red[tid] = u[tid] * W2[tid] + u[tid + 256] * W2[tid + 256];
  __syncthreads();
  for (int s = 128; s > 0; s >>= 1) {
    if (tid < s) red[tid] += red[tid + s];
    __syncthreads();
  }
  if (tid == 0) out[b] = red[0] + b2[0];
}

// ---------------- host launch ----------------
extern "C" void kernel_launch(void* const* d_in, const int* in_sizes, int n_in,
                              void* d_out, int out_size, void* d_ws, size_t ws_size,
                              hipStream_t stream) {
  (void)in_sizes; (void)n_in; (void)out_size; (void)ws_size;

  // setup_inputs() dict order; 'params' flattened in insertion order.
  const float* h0[2]    = { (const float*)d_in[0],  (const float*)d_in[1]  };
  const int*   edg[2]   = { (const int*)  d_in[2],  (const int*)  d_in[3]  };
  const float* eattr    =   (const float*)d_in[4];
  const float* nmask[2] = { (const float*)d_in[5],  (const float*)d_in[8]  };
  const float* emask[2] = { (const float*)d_in[6],  (const float*)d_in[9]  };
  const float* xs[2]    = { (const float*)d_in[11], (const float*)d_in[12] };
  const float* embW[2]  = { (const float*)d_in[13], (const float*)d_in[15] };
  const float* embB[2]  = { (const float*)d_in[14], (const float*)d_in[16] };
  const float* decW1[2] = { (const float*)d_in[17], (const float*)d_in[21] };
  const float* decB1[2] = { (const float*)d_in[18], (const float*)d_in[22] };
  const float* decW2[2] = { (const float*)d_in[19], (const float*)d_in[23] };
  const float* decB2[2] = { (const float*)d_in[20], (const float*)d_in[24] };
  const float* gW1      =   (const float*)d_in[25];
  const float* gB1      =   (const float*)d_in[26];
  const float* gW2      =   (const float*)d_in[27];
  const float* gB2      =   (const float*)d_in[28];
  const float* We1f[2]  = { (const float*)d_in[29], (const float*)d_in[37] };
  const float* be1[2]   = { (const float*)d_in[30], (const float*)d_in[38] };
  const float* We2f[2]  = { (const float*)d_in[31], (const float*)d_in[39] };
  const float* be2[2]   = { (const float*)d_in[32], (const float*)d_in[40] };
  const float* Wn1f[2]  = { (const float*)d_in[33], (const float*)d_in[41] };
  const float* bn1[2]   = { (const float*)d_in[34], (const float*)d_in[42] };
  const float* Wn2f[2]  = { (const float*)d_in[35], (const float*)d_in[43] };
  const float* bn2[2]   = { (const float*)d_in[36], (const float*)d_in[44] };

  // workspace carve-up (256B aligned regions)
  char* ws = (char*)d_ws;
  size_t off = 0;
  auto take = [&](size_t bytes) -> char* {
    char* p = ws + off;
    off += (bytes + 255) & ~(size_t)255;
    return p;
  };

  unsigned short *wWe1[2], *wWe2[2], *wWn1[2], *wWn2[2], *hbf[2];
  float *h[2], *agg[2], *hd[2];
  for (int s = 0; s < 2; ++s) {
    wWe1[s] = (unsigned short*)take((size_t)NL * H * KP * 2);
    wWe2[s] = (unsigned short*)take((size_t)NL * H * H  * 2);
    wWn1[s] = (unsigned short*)take((size_t)NL * H * KP * 2);
    wWn2[s] = (unsigned short*)take((size_t)NL * H * H  * 2);
    h[s]    = (float*)         take((size_t)NNODE * H * 4);
    hbf[s]  = (unsigned short*)take((size_t)NNODE * H * 2);
    agg[s]  = (float*)         take((size_t)NNODE * H * 4);
    hd[s]   = (float*)         take((size_t)NNODE * H * 4);
  }
  float* gsum = (float*)take((size_t)2 * NB * H * 4);

  dim3 blk(256);
  const int gNH = (NNODE * H + 255) / 256;

  zero_kernel<<<(2 * NB * H + 255) / 256, blk, 0, stream>>>(gsum, 2 * NB * H);

  for (int s = 0; s < 2; ++s) {
    wprep_kernel<<<(NL * H * KP + 255) / 256, blk, 0, stream>>>(We1f[s], wWe1[s], D_E1, KP);
    wprep_kernel<<<(NL * H * H  + 255) / 256, blk, 0, stream>>>(We2f[s], wWe2[s], H,    H);
    wprep_kernel<<<(NL * H * KP + 255) / 256, blk, 0, stream>>>(Wn1f[s], wWn1[s], D_N1, KP);
    wprep_kernel<<<(NL * H * H  + 255) / 256, blk, 0, stream>>>(Wn2f[s], wWn2[s], H,    H);
    emb_kernel<<<gNH, blk, 0, stream>>>(h0[s], embW[s], embB[s], h[s]);
  }

  for (int l = 0; l < NL; ++l) {
    for (int s = 0; s < 2; ++s) {
      cvt_zero_kernel<<<gNH, blk, 0, stream>>>(h[s], hbf[s], agg[s]);
      edge_mlp_kernel<<<NE / EM, blk, SMEM_EDGE, stream>>>(
          hbf[s], xs[s], edg[s], eattr, emask[s],
          wWe1[s] + (size_t)l * H * KP, be1[s] + l * H,
          wWe2[s] + (size_t)l * H * H,  be2[s] + l * H,
          agg[s]);
      node_mlp_kernel<<<NNODE / 16, blk, 0, stream>>>(
          hbf[s], agg[s], h0[s],
          wWn1[s] + (size_t)l * H * KP, bn1[s] + l * H,
          wWn2[s] + (size_t)l * H * H,  bn2[s] + l * H,
          h[s]);
    }
  }

  for (int s = 0; s < 2; ++s) {
    dec1_kernel<<<gNH, blk, 0, stream>>>(h[s], decW1[s], decB1[s], hd[s]);
    dec2_kernel<<<gNH, blk, 0, stream>>>(hd[s], decW2[s], decB2[s], nmask[s],
                                         gsum + (size_t)s * NB * H);
  }

  graph_dec_kernel<<<NB, blk, 0, stream>>>(gsum, gsum + (size_t)NB * H,
                                           gW1, gB1, gW2, gB2, (float*)d_out);
}
